// DecoderAttnLSTM_48550310314479
// MI455X (gfx1250) — compile-verified, matrix operand order checked by reading, not run on hardware
//
#include <hip/hip_runtime.h>
#include <hip/hip_bf16.h>

// ---------------------------------------------------------------------------
// Problem dimensions (from reference)
// ---------------------------------------------------------------------------
#define V_    32000
#define E_    512
#define H_    1024      // DEC_H
#define A_    128
#define ENC2_ 1024      // 2*ENC_H
#define B_    32
#define T_    128
#define S_    256
#define KCAT  (E_ + ENC2_ + H_)   // 2560 : [embedded | context | h_prev]
#define G4    (4 * H_)            // 4096 gates

typedef __bf16 bf16;
typedef __attribute__((ext_vector_type(16))) __bf16 v16bf;
typedef __attribute__((ext_vector_type(8)))  __bf16 v8bf;
typedef __attribute__((ext_vector_type(8)))  float  v8f;

// ---------------------------------------------------------------------------
// WMMA fragment load for 16-bit A/B (CDNA5 layout, CDNA5 ISA):
//   lane group half = lane/16; values j=0..7  -> K = k0 + 8*half + j
//                              values j=8..15 -> K = k0 + 8*half + 16 + (j-8)
// i.e. two contiguous 16-byte chunks per lane -> two global_load_b128.
// ---------------------------------------------------------------------------
__device__ __forceinline__ v16bf load_frag(const bf16* __restrict__ base, int ld,
                                           int row, int k0, int half) {
  const bf16* p = base + (size_t)row * ld + k0 + half * 8;
  v8bf lo = *reinterpret_cast<const v8bf*>(p);
  v8bf hi = *reinterpret_cast<const v8bf*>(p + 16);
  return __builtin_shufflevector(lo, hi, 0, 1, 2, 3, 4, 5, 6, 7,
                                 8, 9, 10, 11, 12, 13, 14, 15);
}

__device__ __forceinline__ v8f wmma_bf16(v16bf a, v16bf b, v8f c) {
  return __builtin_amdgcn_wmma_f32_16x16x32_bf16(false, a, false, b,
                                                 (short)0, c, false, false);
}

// ---------------------------------------------------------------------------
// Small-M GEMM: D(32 x N) = A(32 x K, bf16) * B(N x K, bf16)^T + bias
// One wave owns a 32x32 output tile (2 M-tiles x 2 N-tiles, 4 accumulators):
// B fragments shared across both M tiles -> ~32 FLOP/byte of L2 traffic.
// Software-pipelined K loop (peeled last iteration) so next-slice loads issue
// before the current WMMAs; unconditional near-scope prefetch on the B stream.
// N, K multiples of 32. Block = 128 threads (4 waves -> 128 output columns).
// ---------------------------------------------------------------------------
template <bool NT_STORE>
__global__ void wmma_gemm_kernel(const bf16* __restrict__ A,
                                 const bf16* __restrict__ Bm,
                                 const float* __restrict__ bias,
                                 float* __restrict__ D,
                                 int N, int K, long ldD) {
  const int wave = (blockIdx.x * blockDim.x + threadIdx.x) >> 5;
  const int lane = threadIdx.x & 31;
  const int half = lane >> 4;
  const int l16  = lane & 15;
  const int n0   = wave * 32;           // wave-uniform -> EXEC stays all-ones
  if (n0 >= N) return;

  const int bn0 = n0 + l16;
  const int bn1 = n0 + 16 + l16;

  v8f acc00 = {}, acc01 = {}, acc10 = {}, acc11 = {};

  // prologue: fragments for k0 = 0
  v16bf a0 = load_frag(A,  K, l16,      0, half);
  v16bf a1 = load_frag(A,  K, 16 + l16, 0, half);
  v16bf b0 = load_frag(Bm, K, bn0,      0, half);
  v16bf b1 = load_frag(Bm, K, bn1,      0, half);

  for (int k0 = 0; k0 < K - 32; k0 += 32) {
    const int kn = k0 + 32;
    // prefetch the B stream two slices ahead (near-temporal)
    __builtin_prefetch(Bm + (size_t)bn0 * K + k0 + 64, 0, 3);
    __builtin_prefetch(Bm + (size_t)bn1 * K + k0 + 64, 0, 3);
    // issue next-slice loads before consuming current fragments
    v16bf na0 = load_frag(A,  K, l16,      kn, half);
    v16bf na1 = load_frag(A,  K, 16 + l16, kn, half);
    v16bf nb0 = load_frag(Bm, K, bn0,      kn, half);
    v16bf nb1 = load_frag(Bm, K, bn1,      kn, half);
    acc00 = wmma_bf16(a0, b0, acc00);
    acc10 = wmma_bf16(a1, b0, acc10);
    acc01 = wmma_bf16(a0, b1, acc01);
    acc11 = wmma_bf16(a1, b1, acc11);
    a0 = na0; a1 = na1; b0 = nb0; b1 = nb1;
  }
  // peeled last K-slice
  acc00 = wmma_bf16(a0, b0, acc00);
  acc10 = wmma_bf16(a1, b0, acc10);
  acc01 = wmma_bf16(a0, b1, acc01);
  acc11 = wmma_bf16(a1, b1, acc11);

  // C/D layout: M = r + 8*half (+16 for second M tile), N = lane%16
  const int mb = 8 * half;
  const float bv0 = bias ? bias[bn0] : 0.0f;
  const float bv1 = bias ? bias[bn1] : 0.0f;
#pragma unroll
  for (int r = 0; r < 8; ++r) {
    float* p00 = D + (long)(mb + r) * ldD + bn0;
    float* p01 = D + (long)(mb + r) * ldD + bn1;
    float* p10 = D + (long)(mb + 16 + r) * ldD + bn0;
    float* p11 = D + (long)(mb + 16 + r) * ldD + bn1;
    if (NT_STORE) {
      __builtin_nontemporal_store(acc00[r] + bv0, p00);
      __builtin_nontemporal_store(acc01[r] + bv1, p01);
      __builtin_nontemporal_store(acc10[r] + bv0, p10);
      __builtin_nontemporal_store(acc11[r] + bv1, p11);
    } else {
      *p00 = acc00[r] + bv0;
      *p01 = acc01[r] + bv1;
      *p10 = acc10[r] + bv0;
      *p11 = acc11[r] + bv1;
    }
  }
}

// ---------------------------------------------------------------------------
// Per-step attention + x_cat packing. One block per batch row (256 threads).
//   x_cat[b] = [ emb[tok] (512) | context (1024) | h_prev (1024) ]  as bf16
// ---------------------------------------------------------------------------
__global__ void attn_step_kernel(const float* __restrict__ h_prev,   // (B,H)
                                 const float* __restrict__ enc_proj, // (B,S,A)
                                 const float* __restrict__ enc,      // (B,S,ENC2)
                                 const float* __restrict__ Ws,       // (A,H)
                                 const float* __restrict__ vvec,     // (A)
                                 const float* __restrict__ emb,      // (V,E)
                                 const int*   __restrict__ tgt,      // (B,T)
                                 int t,
                                 bf16* __restrict__ x_cat) {         // (B,KCAT)
  const int b   = blockIdx.x;
  const int tid = threadIdx.x;            // 256 threads
  __shared__ float dp[A_];
  __shared__ float sc[S_];
  __shared__ float red[256];

  // embedding gather + h_prev copy into x_cat (independent of attention math)
  const int tok = tgt[b * T_ + (t - 1)];
  for (int e = tid; e < E_; e += 256)
    x_cat[b * KCAT + e] = (bf16)emb[(long)tok * E_ + e];
  for (int j = tid; j < H_; j += 256)
    x_cat[b * KCAT + (E_ + ENC2_) + j] = (bf16)h_prev[b * H_ + j];

  // dec_proj = h_prev @ Ws^T   (A_=128 dots of length H_)
  if (tid < A_) {
    float s = 0.f;
    const float* hr = h_prev + b * H_;
    const float* wr = Ws + tid * H_;
    for (int j = 0; j < H_; ++j) s = fmaf(hr[j], wr[j], s);
    dp[tid] = s;
  }
  __syncthreads();

  // score[s] = sum_a v[a] * tanh(dp[a] + enc_proj[b,s,a])
  {
    const float* ep = enc_proj + ((long)b * S_ + tid) * A_;
    float s = 0.f;
    for (int a = 0; a < A_; ++a) s = fmaf(vvec[a], tanhf(dp[a] + ep[a]), s);
    sc[tid] = s;
  }
  __syncthreads();

  // softmax over S_ = 256
  red[tid] = sc[tid];
  __syncthreads();
  for (int off = 128; off > 0; off >>= 1) {
    if (tid < off) red[tid] = fmaxf(red[tid], red[tid + off]);
    __syncthreads();
  }
  const float m = red[0];
  __syncthreads();
  const float ex = __expf(sc[tid] - m);
  red[tid] = ex;
  __syncthreads();
  for (int off = 128; off > 0; off >>= 1) {
    if (tid < off) red[tid] += red[tid + off];
    __syncthreads();
  }
  const float inv = 1.0f / red[0];
  __syncthreads();
  sc[tid] = ex * inv;
  __syncthreads();

  // context[e] = sum_s attn[s] * enc[b,s,e]  -> bf16 into x_cat
  const float* eb = enc + (long)b * S_ * ENC2_;
  for (int e = tid; e < ENC2_; e += 256) {
    float c = 0.f;
    for (int s = 0; s < S_; ++s) c = fmaf(sc[s], eb[(long)s * ENC2_ + e], c);
    x_cat[b * KCAT + E_ + e] = (bf16)c;
  }
}

// ---------------------------------------------------------------------------
// LSTM elementwise (PyTorch gate order i,f,g,o); writes h as f32 and bf16.
// ---------------------------------------------------------------------------
__global__ void lstm_kernel(const float* __restrict__ gates,   // (B,4H)
                            const float* __restrict__ c_prev,  // (B,H)
                            float* __restrict__ h_new,
                            float* __restrict__ c_new,
                            bf16* __restrict__ hbf) {
  const int idx = blockIdx.x * blockDim.x + threadIdx.x;  // B_*H_
  const int b = idx >> 10, j = idx & (H_ - 1);
  const float* g = gates + (long)b * G4;
  const float gi = g[j], gf = g[j + H_], gg = g[j + 2 * H_], go = g[j + 3 * H_];
  const float si = 1.f / (1.f + __expf(-gi));
  const float sf = 1.f / (1.f + __expf(-gf));
  const float so = 1.f / (1.f + __expf(-go));
  const float cn = sf * c_prev[idx] + si * tanhf(gg);
  const float hn = so * tanhf(cn);
  c_new[idx] = cn;
  h_new[idx] = hn;
  hbf[idx]   = (bf16)hn;
}

// ---------------------------------------------------------------------------
// One-time prep kernels
// ---------------------------------------------------------------------------
__global__ void f2bf_kernel(const float* __restrict__ src, bf16* __restrict__ dst, long n) {
  const long i = (long)blockIdx.x * blockDim.x + threadIdx.x;
  if (i < n) dst[i] = (bf16)src[i];
}

__global__ void wcat_kernel(const float* __restrict__ Wih,   // (4H, E+ENC2)
                            const float* __restrict__ Whh,   // (4H, H)
                            bf16* __restrict__ Wcat) {       // (4H, KCAT)
  const long i = (long)blockIdx.x * blockDim.x + threadIdx.x;
  if (i >= (long)G4 * KCAT) return;
  const int r = (int)(i / KCAT), k = (int)(i % KCAT);
  const float x = (k < E_ + ENC2_) ? Wih[(long)r * (E_ + ENC2_) + k]
                                   : Whh[(long)r * H_ + (k - (E_ + ENC2_))];
  Wcat[i] = (bf16)x;
}

__global__ void encproj_kernel(const float* __restrict__ enc,  // (B*S, ENC2)
                               const float* __restrict__ Wh,   // (A, ENC2)
                               float* __restrict__ ep) {       // (B*S, A)
  const int bs = blockIdx.x;    // B_*S_ blocks
  const int a  = threadIdx.x;   // A_ threads
  const float* er = enc + (long)bs * ENC2_;
  const float* wr = Wh + (long)a * ENC2_;
  float s = 0.f;
  for (int e = 0; e < ENC2_; ++e) s = fmaf(er[e], wr[e], s);
  ep[(long)bs * A_ + a] = s;
}

__global__ void inithc_kernel(const float* __restrict__ hidden,
                              const float* __restrict__ cell,
                              float* __restrict__ h0, float* __restrict__ c0) {
  const int i = blockIdx.x * blockDim.x + threadIdx.x;
  if (i < B_ * H_) { h0[i] = hidden[i]; c0[i] = cell[i]; }
}

__global__ void bsum_kernel(const float* __restrict__ bih,
                            const float* __restrict__ bhh,
                            float* __restrict__ bsum) {
  const int i = blockIdx.x * blockDim.x + threadIdx.x;
  if (i < G4) bsum[i] = bih[i] + bhh[i];
}

__global__ void zero0_kernel(float* __restrict__ out) {  // out[:,0,:] = 0
  const long i = (long)blockIdx.x * blockDim.x + threadIdx.x;
  if (i < (long)B_ * V_) {
    const long b = i / V_, n = i % V_;
    __builtin_nontemporal_store(0.0f, out + b * (long)T_ * V_ + n);
  }
}

// ---------------------------------------------------------------------------
// Host: graph-capture-safe launch sequence (all on `stream`, no sync/alloc)
// ---------------------------------------------------------------------------
extern "C" void kernel_launch(void* const* d_in, const int* in_sizes, int n_in,
                              void* d_out, int out_size, void* d_ws, size_t ws_size,
                              hipStream_t stream) {
  const int*   tgt    = (const int*)  d_in[0];
  const float* enc    = (const float*)d_in[1];
  const float* hidden = (const float*)d_in[2];
  const float* cell   = (const float*)d_in[3];
  const float* emb    = (const float*)d_in[4];
  const float* Wih    = (const float*)d_in[5];
  const float* Whh    = (const float*)d_in[6];
  const float* bih    = (const float*)d_in[7];
  const float* bhh    = (const float*)d_in[8];
  const float* Ws     = (const float*)d_in[9];
  const float* Wh     = (const float*)d_in[10];
  const float* vvec   = (const float*)d_in[11];
  const float* fcW    = (const float*)d_in[12];
  const float* fcb    = (const float*)d_in[13];
  float* out = (float*)d_out;

  // workspace carve-up (~92 MB)
  char* w = (char*)d_ws;
  bf16*  fcWb = (bf16*)w;  w += (size_t)V_ * H_ * sizeof(bf16);      // 65.5 MB
  bf16*  Wcat = (bf16*)w;  w += (size_t)G4 * KCAT * sizeof(bf16);    // 21.0 MB
  float* encp = (float*)w; w += (size_t)B_ * S_ * A_ * sizeof(float);
  bf16*  xcat = (bf16*)w;  w += (size_t)B_ * KCAT * sizeof(bf16);
  float* gates = (float*)w; w += (size_t)B_ * G4 * sizeof(float);
  float* hbuf0 = (float*)w; w += (size_t)B_ * H_ * sizeof(float);
  float* hbuf1 = (float*)w; w += (size_t)B_ * H_ * sizeof(float);
  float* cbuf0 = (float*)w; w += (size_t)B_ * H_ * sizeof(float);
  float* cbuf1 = (float*)w; w += (size_t)B_ * H_ * sizeof(float);
  bf16*  hbf  = (bf16*)w;  w += (size_t)B_ * H_ * sizeof(bf16);
  float* bsum = (float*)w; w += (size_t)G4 * sizeof(float);
  float* hb[2] = {hbuf0, hbuf1};
  float* cb[2] = {cbuf0, cbuf1};

  // --- one-time prep ---
  { long n = (long)V_ * H_;
    f2bf_kernel<<<(unsigned)((n + 255) / 256), 256, 0, stream>>>(fcW, fcWb, n); }
  { long n = (long)G4 * KCAT;
    wcat_kernel<<<(unsigned)((n + 255) / 256), 256, 0, stream>>>(Wih, Whh, Wcat); }
  encproj_kernel<<<B_ * S_, A_, 0, stream>>>(enc, Wh, encp);
  inithc_kernel<<<(B_ * H_ + 255) / 256, 256, 0, stream>>>(hidden, cell, hb[0], cb[0]);
  bsum_kernel<<<(G4 + 255) / 256, 256, 0, stream>>>(bih, bhh, bsum);
  { long n = (long)B_ * V_;
    zero0_kernel<<<(unsigned)((n + 255) / 256), 256, 0, stream>>>(out); }

  // --- 127 sequential decode steps ---
  // one wave per 32 output columns; 4 waves (128 cols) per block
  const int gatesBlocks  = G4 / 128;   // 32
  const int logitsBlocks = V_ / 128;   // 250
  for (int t = 1; t < T_; ++t) {
    const float* hp = hb[(t - 1) & 1];
    const float* cp = cb[(t - 1) & 1];
    float* hn = hb[t & 1];
    float* cn = cb[t & 1];

    attn_step_kernel<<<B_, 256, 0, stream>>>(hp, encp, enc, Ws, vvec, emb, tgt, t, xcat);
    wmma_gemm_kernel<false><<<gatesBlocks, 128, 0, stream>>>(xcat, Wcat, bsum, gates,
                                                             G4, KCAT, (long)G4);
    lstm_kernel<<<(B_ * H_) / 256, 256, 0, stream>>>(gates, cp, hn, cn, hbf);
    wmma_gemm_kernel<true><<<logitsBlocks, 128, 0, stream>>>(hbf, fcWb, fcb,
                                                             out + (long)t * V_,
                                                             V_, H_, (long)T_ * V_);
  }
}
